// ForestCLF_8014408974383
// MI455X (gfx1250) — matrix operand order repaired and checked by reading, not compiled, not run on hardware
//
#include <hip/hip_runtime.h>

typedef __attribute__((ext_vector_type(16))) _Float16 v16h;
typedef __attribute__((ext_vector_type(8)))  float    v8f;

#define B_      1024
#define T_      64
#define F_      1024
#define C_      3
#define N2_     16
#define NSLOT_  5
#define KC_     32            // K-chunks of 32 covering F=1024
#define TPW_    8             // trees per workgroup

// workspace layout (bytes)
#define ACC_FLOATS (27 * B_)                 // A5(5B) C5(5B) A16(16B) GS(B)
#define OFF_WA     (112 * 1024)              // f16 combined w_clc|w1 (swizzled)
#define OFF_WB     (OFF_WA + 2 * 1024 * 1024) // f16 w2[0:1024] (swizzled)

// ---------------------------------------------------------------------------
// Zero the accumulator region (must run every launch; harness poisons ws).
__global__ void zero_acc(float* __restrict__ acc) {
    int i = blockIdx.x * blockDim.x + threadIdx.x;
    if (i < ACC_FLOATS) acc[i] = 0.0f;
}

// ---------------------------------------------------------------------------
// Convert weights to f16 in WMMA B-fragment lane order.
// Contiguous-K order used here, same as the A gather (16-bit B, 32x16): lane L owns column n = L%16;
// lanes 0-15 hold K = kc*32 + 0..15, lanes 16-31 hold K = kc*32 + 16..31,
// element e of the v16h = K offset e.  One 32B load per lane in the main loop.
__global__ void prep_weights(const float* __restrict__ w_clc,
                             const float* __restrict__ w1,
                             const float* __restrict__ w2,
                             _Float16* __restrict__ WA,
                             _Float16* __restrict__ WB) {
    int tid = blockIdx.x * blockDim.x + threadIdx.x;      // (t, kc, lane)
    if (tid >= T_ * KC_ * 32) return;
    int lane = tid & 31;
    int kc   = (tid >> 5) & 31;
    int t    = tid >> 10;
    int n    = lane & 15;
    int kb   = kc * 32 + ((lane >> 4) << 4);
    v16h a, b;
#pragma unroll
    for (int e = 0; e < 16; ++e) {
        int k = kb + e;
        float wav = (n < 2) ? w_clc[((size_t)t * F_ + k) * 2 + n]
                  : (n < 5) ? w1[((size_t)t * F_ + k) * 3 + (n - 2)]
                  : 0.0f;
        a[e] = (_Float16)wav;
        b[e] = (_Float16)w2[((size_t)t * (F_ + C_) + k) * N2_ + n];
    }
    *(v16h*)(WA + (size_t)tid * 16) = a;
    *(v16h*)(WB + (size_t)tid * 16) = b;
}

// ---------------------------------------------------------------------------
// Main kernel: one wave (32 threads) owns a 16-row batch tile and TPW_ trees.
// Per tree: gather A fragments from L2-resident d via LDS-staged idx, run
// 2 x 32 v_wmma_f32_16x16x32_f16 (K=1024), epilogue through a 16x16 LDS tile.
__global__ void __launch_bounds__(32)
forest_main(const float* __restrict__ d,   const int* __restrict__ idx,
            const int*   __restrict__ cc,
            const float* __restrict__ b_clc, const float* __restrict__ b1,
            const float* __restrict__ w2,    const float* __restrict__ b2,
            const _Float16* __restrict__ WA, const _Float16* __restrict__ WB,
            float* __restrict__ accbuf) {
    __shared__ int   s_idx[F_];
    __shared__ float s_tile[16 * 16];
    __shared__ float s_relu[16 * 3];
    __shared__ float s_gate[16];

    const int lane   = threadIdx.x;
    const int b0     = blockIdx.x * 16;
    const int t0     = blockIdx.y * TPW_;
    const int n      = lane & 15;
    const int mA     = (lane >> 4) << 3;   // C-frag row base: 0 or 8
    const int kbaseA = (lane >> 4) << 4;   // A-frag K base:  0 or 16
    const int row    = b0 + (lane & 15);
    const float* drow = d + (size_t)row * 2048;

    float acc5[NSLOT_] = {}, cnt5[NSLOT_] = {}, acc16[N2_] = {};
    float gsum = 0.0f;

    for (int tt = 0; tt < TPW_; ++tt) {
        const int t = t0 + tt;

        // stage this tree's gather indices (reused by all 32 K-chunks)
#pragma unroll 4
        for (int j = 0; j < 32; ++j)
            s_idx[j * 32 + lane] = idx[(size_t)t * F_ + j * 32 + lane];
        __syncthreads();

        v8f accA = {}; v8f accB = {};
        const _Float16* wa_t = WA + ((size_t)t * KC_ * 32 + lane) * 16;
        const _Float16* wb_t = WB + ((size_t)t * KC_ * 32 + lane) * 16;

        for (int kc = 0; kc < KC_; ++kc) {
            const int kb = kc * 32 + kbaseA;
            v16h af;
#pragma unroll
            for (int j = 0; j < 8; ++j) {        // contiguous K layout (matches WA/WB)
                af[j]     = (_Float16)drow[s_idx[kb + j]];
                af[8 + j] = (_Float16)drow[s_idx[kb + 8 + j]];
            }
            v16h wa = *(const v16h*)(wa_t + (size_t)kc * 512);
            v16h wb = *(const v16h*)(wb_t + (size_t)kc * 512);
            accA = __builtin_amdgcn_wmma_f32_16x16x32_f16(
                       false, af, false, wa, (short)0, accA, false, false);
            accB = __builtin_amdgcn_wmma_f32_16x16x32_f16(
                       false, af, false, wb, (short)0, accB, false, false);
        }

        // L_A (clc|logits1) fragment -> LDS tile
#pragma unroll
        for (int r = 0; r < 8; ++r) s_tile[(mA + r) * 16 + n] = accA[r];
        __syncthreads();

        if (lane < 16) {                          // one batch row per lane
            const int m = lane;
            float lc0 = s_tile[m * 16 + 0] + b_clc[t * 2 + 0];
            float lc1 = s_tile[m * 16 + 1] + b_clc[t * 2 + 1];
            float mx  = fmaxf(lc0, lc1);
            float e0  = __expf(lc0 - mx), e1 = __expf(lc1 - mx);
            float gate = e1 / (e0 + e1);          // exp(log_softmax[...,1])
            s_gate[m] = gate;

            float l1v[3];
#pragma unroll
            for (int c = 0; c < 3; ++c) l1v[c] = s_tile[m * 16 + 2 + c] + b1[t * 3 + c];
            float m1 = fmaxf(fmaxf(l1v[0], l1v[1]), l1v[2]);
            float p[3], ps = 0.0f;
#pragma unroll
            for (int c = 0; c < 3; ++c) { p[c] = __expf(l1v[c] - m1); ps += p[c]; }
            float inv = 1.0f / ps;
#pragma unroll
            for (int c = 0; c < 3; ++c) {
                p[c] *= inv;
                s_relu[m * 3 + c] = fmaxf(l1v[c], 0.0f);
                int slot = cc[t * 3 + c];
                float pg = p[c] * gate;
#pragma unroll
                for (int s = 0; s < NSLOT_; ++s) {   // keep arrays in VGPRs
                    acc5[s] += (slot == s) ? pg   : 0.0f;
                    cnt5[s] += (slot == s) ? gate : 0.0f;
                }
            }
            gsum += gate;
        }
        __syncthreads();

        // rank-3 relu tail of w2 + bias, write L2 tile
        float wt0 = w2[((size_t)t * (F_ + C_) + F_ + 0) * N2_ + n];
        float wt1 = w2[((size_t)t * (F_ + C_) + F_ + 1) * N2_ + n];
        float wt2 = w2[((size_t)t * (F_ + C_) + F_ + 2) * N2_ + n];
        float bb  = b2[t * N2_ + n];
#pragma unroll
        for (int r = 0; r < 8; ++r) {
            const int m = mA + r;
            float v = accB[r] + bb
                    + s_relu[m * 3 + 0] * wt0
                    + s_relu[m * 3 + 1] * wt1
                    + s_relu[m * 3 + 2] * wt2;
            s_tile[m * 16 + n] = v;
        }
        __syncthreads();

        if (lane < 16) {                          // softmax over 16, accumulate
            const int m = lane;
            float mx2 = -1e30f;
#pragma unroll
            for (int j = 0; j < 16; ++j) mx2 = fmaxf(mx2, s_tile[m * 16 + j]);
            float e[16], ss = 0.0f;
#pragma unroll
            for (int j = 0; j < 16; ++j) { e[j] = __expf(s_tile[m * 16 + j] - mx2); ss += e[j]; }
            float g_ = s_gate[m] / ss;
#pragma unroll
            for (int j = 0; j < 16; ++j) acc16[j] += e[j] * g_;
        }
        __syncthreads();
    }

    // flush per-row accumulators (cross tree-group reduction)
    if (lane < 16) {
        const int b = b0 + lane;
        float* A5  = accbuf;
        float* C5  = accbuf + 5 * B_;
        float* A16 = accbuf + 10 * B_;
        float* GS  = accbuf + 26 * B_;
#pragma unroll
        for (int s = 0; s < NSLOT_; ++s) {
            atomicAdd(&A5[b * 5 + s], acc5[s]);
            atomicAdd(&C5[b * 5 + s], cnt5[s]);
        }
#pragma unroll
        for (int j = 0; j < N2_; ++j) atomicAdd(&A16[b * 16 + j], acc16[j]);
        atomicAdd(&GS[b], gsum);
    }
}

// ---------------------------------------------------------------------------
__global__ void finalize(const float* __restrict__ accbuf, float* __restrict__ out) {
    int b = blockIdx.x * blockDim.x + threadIdx.x;
    if (b >= B_) return;
    const float* A5  = accbuf;
    const float* C5  = accbuf + 5 * B_;
    const float* A16 = accbuf + 10 * B_;
    const float* GS  = accbuf + 26 * B_;
    float inv_g = 1.0f / GS[b];
#pragma unroll
    for (int s = 0; s < NSLOT_; ++s) {
        float c = C5[b * 5 + s];
        out[b * 21 + s] = (c > 0.0f) ? A5[b * 5 + s] / c : 0.0f;
    }
#pragma unroll
    for (int j = 0; j < N2_; ++j) out[b * 21 + 5 + j] = A16[b * 16 + j] * inv_g;
}

// ---------------------------------------------------------------------------
extern "C" void kernel_launch(void* const* d_in, const int* in_sizes, int n_in,
                              void* d_out, int out_size, void* d_ws, size_t ws_size,
                              hipStream_t stream) {
    const float* d_d   = (const float*)d_in[0];
    const int*   d_idx = (const int*)  d_in[1];
    const int*   d_cc  = (const int*)  d_in[2];
    const float* w_clc = (const float*)d_in[3];
    const float* b_clc = (const float*)d_in[4];
    const float* w1    = (const float*)d_in[5];
    const float* b1    = (const float*)d_in[6];
    const float* w2    = (const float*)d_in[7];
    const float* b2    = (const float*)d_in[8];
    float* out = (float*)d_out;

    char* ws = (char*)d_ws;
    float*    accbuf = (float*)ws;
    _Float16* WA     = (_Float16*)(ws + OFF_WA);
    _Float16* WB     = (_Float16*)(ws + OFF_WB);

    zero_acc<<<(ACC_FLOATS + 255) / 256, 256, 0, stream>>>(accbuf);
    prep_weights<<<(T_ * KC_ * 32 + 255) / 256, 256, 0, stream>>>(w_clc, w1, w2, WA, WB);
    dim3 grid(B_ / 16, T_ / TPW_);
    forest_main<<<grid, 32, 0, stream>>>(d_d, d_idx, d_cc, b_clc, b1, w2, b2, WA, WB, accbuf);
    finalize<<<(B_ + 255) / 256, 256, 0, stream>>>(accbuf, out);
}
